// LSTM_Net_86723979640964
// MI455X (gfx1250) — matrix operand order repaired and checked by reading, not compiled, not run on hardware
//
#include <hip/hip_runtime.h>
#include <hip/hip_bf16.h>

// ---------------------------------------------------------------------------
// Problem constants (from reference)
// ---------------------------------------------------------------------------
#define BB 64
#define TT 256
#define IN_DIM 64
#define HH 768
#define G4 (4 * HH)        // 3072 gates
#define LU 256
#define BT (BB * TT)       // 16384

typedef __attribute__((ext_vector_type(16))) __bf16 v16bf;
typedef __attribute__((ext_vector_type(8)))  float  v8f;

union Tile32 { uint4 q[2]; v16bf v; };

// A-matrix 16x32 bf16 tile load (row_base -> element [tile_row0][k0]), ld in elems.
// lane m<16: elems0-7 = K0..7, elems8-15 = K16..23 ; lane m+16: K8..15 / K24..31
__device__ __forceinline__ v16bf load_a_tile(const __hip_bfloat16* row_base, int ld, int lane) {
    const int half = lane >> 4, m = lane & 15;
    const char* p = (const char*)(row_base + (size_t)m * ld);
    Tile32 t;
    t.q[0] = *(const uint4*)(p + 16 * half);
    t.q[1] = *(const uint4*)(p + 32 + 16 * half);
    return t.v;
}

// B-matrix 32x16 bf16 tile load from B^T stored row-major (row n = column n of B).
// lane n<16: elems0-7 = K0..7, elems8-15 = K16..23 ; lane n+16: K8..15 / K24..31
__device__ __forceinline__ v16bf load_b_tile(const __hip_bfloat16* row_base, int ld, int lane) {
    const int half = lane >> 4, n = lane & 15;
    const char* p = (const char*)(row_base + (size_t)n * ld);
    Tile32 t;
    t.q[0] = *(const uint4*)(p + 16 * half);
    t.q[1] = *(const uint4*)(p + 32 + 16 * half);
    return t.v;
}

__device__ __forceinline__ float sigmoidf_(float x) { return 1.0f / (1.0f + __expf(-x)); }

// ---------------------------------------------------------------------------
// Accumulate acc[0..3] (+=) with A[16 x K] @ 4 W-rowblocks.
// K is compile-time: fully unrolled straight-line body lets the scheduler
// hoist loads across iterations with NO register-rotation copies.
// W rowblock q starts at W + q*wstep (each block = 16 rows of ldw elems).
// ---------------------------------------------------------------------------
template <int K>
__device__ __forceinline__ void wmma_rowblock4(v8f acc[4],
        const __hip_bfloat16* __restrict__ A, int lda,
        const __hip_bfloat16* __restrict__ W, size_t wstep, int ldw,
        int lane)
{
    #pragma unroll
    for (int kk = 0; kk < K; kk += 32) {
        const v16bf a = load_a_tile(A + kk, lda, lane);
        #pragma unroll
        for (int q = 0; q < 4; ++q) {
            const v16bf b = load_b_tile(W + q * wstep + kk, ldw, lane);
            acc[q] = __builtin_amdgcn_wmma_f32_16x16x32_bf16(false, a, false, b,
                                                             (short)0, acc[q], false, false);
        }
    }
}

// i/f/g/o -> cell + hidden update for one 16x16 gate-column block
__device__ __forceinline__ void lstm_cell(const v8f acc[4], float* __restrict__ c,
        __hip_bfloat16* __restrict__ hnext, int mrow, int j, int half, int lm)
{
    #pragma unroll
    for (int r = 0; r < 8; ++r) {
        const int b  = mrow + r + 8 * half;
        const int hc = 16 * j + lm;
        const size_t ci = (size_t)b * HH + hc;
        const float si = sigmoidf_(acc[0][r]);
        const float sf = sigmoidf_(acc[1][r]);
        const float tg = tanhf(acc[2][r]);
        const float so = sigmoidf_(acc[3][r]);
        const float cn = sf * c[ci] + si * tg;
        c[ci] = cn;
        hnext[ci] = __float2bfloat16(so * tanhf(cn));
    }
}

// ---------------------------------------------------------------------------
// Generic GEMM: out[M][N] (f32) = A[M][K](bf16) @ W[N][K]^T(bf16) + bias[N]
// grid = (N/64, M/64), block = 128 threads (4 waves); wave w -> mtile, 4 ntiles.
// ---------------------------------------------------------------------------
template <bool SOFTSIGN, int K>
__global__ void gemm_bias_act(const __hip_bfloat16* __restrict__ A,
                              const __hip_bfloat16* __restrict__ W,
                              const float* __restrict__ bias,
                              float* __restrict__ out,
                              int M, int N) {
    const int lane = threadIdx.x & 31;
    const int wave = threadIdx.x >> 5;
    const int mrow = (blockIdx.y * 4 + wave) * 16;
    const int nbase = blockIdx.x * 64;
    const int half = lane >> 4, lm = lane & 15;

    v8f acc[4];
    for (int q = 0; q < 4; ++q) {
        const float bv = bias ? bias[nbase + q * 16 + lm] : 0.0f;
        for (int r = 0; r < 8; ++r) acc[q][r] = bv;
    }
    wmma_rowblock4<K>(acc, A + (size_t)mrow * K, K,
                      W + (size_t)nbase * K, (size_t)16 * K, K, lane);
    for (int q = 0; q < 4; ++q)
        for (int r = 0; r < 8; ++r) {
            float v = acc[q][r];
            if (SOFTSIGN) v = v / (1.0f + fabsf(v));
            out[(size_t)(mrow + r + 8 * half) * N + nbase + q * 16 + lm] = v;
        }
}

// ---------------------------------------------------------------------------
// Pipelined two-layer LSTM step. grid = 96 blocks (48 per layer), 128 threads.
// Blocks 0..47   : layer-0 step t     (gates = xg0[t] + h0[t-1] @ Whh0^T)
// Blocks 48..95  : layer-1 step t-1   (gates = bias1 + h1[t-1] @ Wih1^T
//                                              + h2[t-2] @ Whh1^T)
// h1[t-1] is simply layer-0's state buffer at parity t&1: read-only for both
// layers this launch, overwritten only at launch t+1 (stream-ordered).
// ---------------------------------------------------------------------------
__global__ void lstm_pipe_step(
        const float* __restrict__ xg0,             // [B*T][4H]
        const __hip_bfloat16* __restrict__ whh0,   // [4H][H]
        const __hip_bfloat16* __restrict__ h0prev, // [B][H]  (== h1[t-1] source)
        __hip_bfloat16* __restrict__ h0next,
        float* __restrict__ c0,
        const __hip_bfloat16* __restrict__ wih1,   // [4H][H]
        const __hip_bfloat16* __restrict__ whh1,   // [4H][H]
        const float* __restrict__ bias1,           // [4H]
        const __hip_bfloat16* __restrict__ h2prev,
        __hip_bfloat16* __restrict__ h2next,
        float* __restrict__ c1,
        int t, int l0on, int l1on)
{
    const int lane = threadIdx.x & 31;
    const int wave = threadIdx.x >> 5;
    const int mrow = wave * 16;
    const int half = lane >> 4, lm = lane & 15;
    const bool is_l1 = blockIdx.x >= (HH / 16);
    const int j = is_l1 ? (int)blockIdx.x - HH / 16 : (int)blockIdx.x;

    v8f acc[4];
    if (!is_l1) {
        if (!l0on) return;
        // accumulators seeded from precomputed xg0 (includes both biases)
        for (int q = 0; q < 4; ++q)
            for (int r = 0; r < 8; ++r) {
                const int b = mrow + r + 8 * half;
                acc[q][r] = xg0[((size_t)b * TT + t) * G4 + q * HH + 16 * j + lm];
            }
        wmma_rowblock4<HH>(acc, h0prev + (size_t)mrow * HH, HH,
                           whh0 + (size_t)(16 * j) * HH, (size_t)HH * HH, HH, lane);
        lstm_cell(acc, c0, h0next, mrow, j, half, lm);
    } else {
        if (!l1on) return;
        for (int q = 0; q < 4; ++q) {
            const float bv = bias1[q * HH + 16 * j + lm];
            for (int r = 0; r < 8; ++r) acc[q][r] = bv;
        }
        // input contribution: h1[t-1] @ Wih1^T  (h1[t-1] lives in h0prev)
        wmma_rowblock4<HH>(acc, h0prev + (size_t)mrow * HH, HH,
                           wih1 + (size_t)(16 * j) * HH, (size_t)HH * HH, HH, lane);
        // recurrent contribution: h2[t-2] @ Whh1^T
        wmma_rowblock4<HH>(acc, h2prev + (size_t)mrow * HH, HH,
                           whh1 + (size_t)(16 * j) * HH, (size_t)HH * HH, HH, lane);
        lstm_cell(acc, c1, h2next, mrow, j, half, lm);
    }
}

// ---------------------------------------------------------------------------
// Utility kernels
// ---------------------------------------------------------------------------
__global__ void k_f32_to_bf16(const float* __restrict__ src, __hip_bfloat16* __restrict__ dst,
                              size_t n, float scale) {
    size_t i = (size_t)blockIdx.x * blockDim.x + threadIdx.x;
    const size_t stride = (size_t)gridDim.x * blockDim.x;
    for (; i < n; i += stride) dst[i] = __float2bfloat16(src[i] * scale);
}

__global__ void k_add_bias(const float* __restrict__ a, const float* __restrict__ b,
                           float* __restrict__ dst, int n) {
    int i = blockIdx.x * blockDim.x + threadIdx.x;
    if (i < n) dst[i] = a[i] + b[i];
}

__global__ void k_zero_state(__hip_bfloat16* __restrict__ h0, __hip_bfloat16* __restrict__ h2,
                             float* __restrict__ c0, float* __restrict__ c1, size_t n) {
    size_t i = (size_t)blockIdx.x * blockDim.x + threadIdx.x;
    const size_t stride = (size_t)gridDim.x * blockDim.x;
    const __hip_bfloat16 zb = __float2bfloat16(0.0f);
    for (; i < n; i += stride) {
        h0[i] = zb; h0[n + i] = zb;
        h2[i] = zb; h2[n + i] = zb;
        c0[i] = 0.0f; c1[i] = 0.0f;
    }
}

__global__ void k_final(const float* __restrict__ z, const float* __restrict__ W2,
                        const float* __restrict__ b2, float* __restrict__ out) {
    const int b = threadIdx.x;
    if (b < BB) {
        float s = 0.0f;
        for (int n = 0; n < LU; ++n) s += z[b * LU + n] * W2[n];
        out[b] = (s + b2[0]) * 70.0f;
    }
}

// ---------------------------------------------------------------------------
// Host launcher
// ---------------------------------------------------------------------------
extern "C" void kernel_launch(void* const* d_in, const int* in_sizes, int n_in,
                              void* d_out, int out_size, void* d_ws, size_t ws_size,
                              hipStream_t stream) {
    const float* x    = (const float*)d_in[0];
    const float* Wih0 = (const float*)d_in[1];
    const float* Whh0 = (const float*)d_in[2];
    const float* bih0 = (const float*)d_in[3];
    const float* bhh0 = (const float*)d_in[4];
    const float* Wih1 = (const float*)d_in[5];
    const float* Whh1 = (const float*)d_in[6];
    const float* bih1 = (const float*)d_in[7];
    const float* bhh1 = (const float*)d_in[8];
    const float* W1   = (const float*)d_in[9];
    const float* b1   = (const float*)d_in[10];
    const float* W2   = (const float*)d_in[11];
    const float* b2   = (const float*)d_in[12];
    float* out = (float*)d_out;

    // workspace carve-up (256B aligned segments)
    char* ws = (char*)d_ws;
    size_t off = 0;
    auto alloc = [&](size_t bytes) -> void* {
        off = (off + 255) & ~(size_t)255;
        void* p = ws + off;
        off += bytes;
        return p;
    };
    __hip_bfloat16* xbf   = (__hip_bfloat16*)alloc((size_t)BT * IN_DIM * 2);
    __hip_bfloat16* wih0b = (__hip_bfloat16*)alloc((size_t)G4 * IN_DIM * 2);
    __hip_bfloat16* whh0b = (__hip_bfloat16*)alloc((size_t)G4 * HH * 2);
    __hip_bfloat16* wih1b = (__hip_bfloat16*)alloc((size_t)G4 * HH * 2);
    __hip_bfloat16* whh1b = (__hip_bfloat16*)alloc((size_t)G4 * HH * 2);
    __hip_bfloat16* w1b   = (__hip_bfloat16*)alloc((size_t)LU * HH * 2);
    float* bias0 = (float*)alloc(G4 * 4);
    float* bias1 = (float*)alloc(G4 * 4);
    float* xg0   = (float*)alloc((size_t)BT * G4 * 4);
    __hip_bfloat16* h0buf = (__hip_bfloat16*)alloc((size_t)2 * BB * HH * 2);
    __hip_bfloat16* h2buf = (__hip_bfloat16*)alloc((size_t)2 * BB * HH * 2);
    float* c0 = (float*)alloc((size_t)BB * HH * 4);
    float* c1 = (float*)alloc((size_t)BB * HH * 4);
    float* z  = (float*)alloc((size_t)BB * LU * 4);

    const size_t S = (size_t)BB * HH;   // one state buffer

    // 1) convert inputs/weights to bf16 (x also folds 1/INPUT_SCALE)
    const dim3 cbk(256);
    k_f32_to_bf16<<<2048, cbk, 0, stream>>>(x, xbf, (size_t)BT * IN_DIM, 1.0f / 1.5f);
    k_f32_to_bf16<<<2048, cbk, 0, stream>>>(Wih0, wih0b, (size_t)G4 * IN_DIM, 1.0f);
    k_f32_to_bf16<<<4096, cbk, 0, stream>>>(Whh0, whh0b, (size_t)G4 * HH, 1.0f);
    k_f32_to_bf16<<<4096, cbk, 0, stream>>>(Wih1, wih1b, (size_t)G4 * HH, 1.0f);
    k_f32_to_bf16<<<4096, cbk, 0, stream>>>(Whh1, whh1b, (size_t)G4 * HH, 1.0f);
    k_f32_to_bf16<<<1024, cbk, 0, stream>>>(W1, w1b, (size_t)LU * HH, 1.0f);
    k_add_bias<<<(G4 + 255) / 256, cbk, 0, stream>>>(bih0, bhh0, bias0, G4);
    k_add_bias<<<(G4 + 255) / 256, cbk, 0, stream>>>(bih1, bhh1, bias1, G4);
    k_zero_state<<<192, cbk, 0, stream>>>(h0buf, h2buf, c0, c1, S);

    // 2) xg0 = (x/1.5) @ Wih0^T + (bih0+bhh0)   [16384 x 3072, K=64]
    gemm_bias_act<false, IN_DIM><<<dim3(G4 / 64, BT / 64), dim3(128), 0, stream>>>(
        xbf, wih0b, bias0, xg0, BT, G4);

    // 3) pipelined scans: launch t runs layer-0 step t and layer-1 step t-1
    for (int t = 0; t <= TT; ++t) {
        const __hip_bfloat16* h0p = h0buf + (size_t)(t & 1) * S;
        __hip_bfloat16*       h0n = h0buf + (size_t)((t + 1) & 1) * S;
        const __hip_bfloat16* h2p = h2buf + (size_t)((t + 1) & 1) * S;  // parity (t-1)&1
        __hip_bfloat16*       h2n = h2buf + (size_t)(t & 1) * S;
        lstm_pipe_step<<<dim3(2 * HH / 16), dim3(128), 0, stream>>>(
            xg0, whh0b, h0p, h0n, c0,
            wih1b, whh1b, bias1, h2p, h2n, c1,
            t, (t < TT) ? 1 : 0, (t >= 1) ? 1 : 0);
    }

    // 4) head: z = softsign(h2_last @ W1^T + b1); last h2 at parity 0 (T even)
    gemm_bias_act<true, HH><<<dim3(LU / 64, BB / 64), dim3(128), 0, stream>>>(
        h2buf, w1b, b1, z, BB, LU);

    // 5) pred = (z @ W2^T + b2) * 70
    k_final<<<1, dim3(64), 0, stream>>>(z, W2, b2, out);
    (void)in_sizes; (void)n_in; (void)out_size; (void)ws_size;
}
